// UnsupervisedGIN_3753801416792
// MI455X (gfx1250) — matrix operand order repaired and checked by reading, not compiled, not run on hardware
//
#include <hip/hip_runtime.h>
#include <hip/hip_bf16.h>

typedef __bf16 bf16_t;
typedef __attribute__((ext_vector_type(16))) __bf16 v16bf;
typedef __attribute__((ext_vector_type(8)))  float  v8f;

#define D      128
#define OUTD   64
#define NGRAPH 64
#define NLAYER 4
#define BN_EPS 1e-5f

// ---------------------------------------------------------------- helpers
__device__ __forceinline__ bf16_t f2bf(float x) {
  union { float f; unsigned u; } v; v.f = x;
  unsigned r = v.u + 0x7FFFu + ((v.u >> 16) & 1u);   // round-to-nearest-even
  unsigned short hs = (unsigned short)(r >> 16);
  union { unsigned short s; bf16_t b; } o; o.s = hs;
  return o.b;
}

// ---------------------------------------------------------------- zero
__global__ void k_zero(float* __restrict__ p, long n) {
  long i = (long)blockIdx.x * blockDim.x + threadIdx.x;
  if (i < n) p[i] = 0.0f;
}

// ------------------------------------------------- edge scatter: agg[dst]+=h[src]
// one wave32 per edge; float4 read per lane, 4 f32 atomics per lane (agg is L2-resident)
__global__ void k_scatter(const float* __restrict__ h, const int* __restrict__ src,
                          const int* __restrict__ dst, float* __restrict__ agg, int E) {
  int e    = (int)(((long)blockIdx.x * blockDim.x + threadIdx.x) >> 5);
  int lane = threadIdx.x & 31;
  if (e >= E) return;
  int s = src[e], d = dst[e];
  const float4* hp = (const float4*)(h + (long)s * D);
  float*        ap = agg + (long)d * D;
  float4 v = hp[lane];
  atomicAdd(&ap[lane * 4 + 0], v.x);
  atomicAdd(&ap[lane * 4 + 1], v.y);
  atomicAdd(&ap[lane * 4 + 2], v.z);
  atomicAdd(&ap[lane * 4 + 3], v.w);
}

// ------------------------------------------------- agg = (1+eps[layer])*h + agg
__global__ void k_axpy(float* __restrict__ agg, const float* __restrict__ h,
                       const float* __restrict__ eps, int layer, long n) {
  long i = (long)blockIdx.x * blockDim.x + threadIdx.x;
  if (i >= n) return;
  float e = 1.0f + eps[layer];
  agg[i] = e * h[i] + agg[i];
}

// ------------------------------------------------- WMMA GEMM: C = A @ W + bias
// A: [nRows x 128] f32 (row-major), W: [128 x 128] f32, C: [nRows x 128] f32.
// Block = 8 waves; wave w owns rows [blk*128 + 16w, +16); 8 col-tiles of 16.
// W is converted to bf16 and staged in LDS pre-swizzled per lane as WMMA B fragments in contiguous k order:
//   Bs[((kstep*8 + ntile)*32 + lane)*16 + e] = W[kstep*32 + 16*(lane>>4) + e][ntile*16 + (lane&15)]
// Epilogue additionally accumulates column sums / sum-of-squares of C (post-bias)
// for the following BatchNorm: block-local LDS reduce (ds_add_f32), then one
// global f32 atomic per column per block into gstats[0..255].
__global__ __launch_bounds__(256) void k_gemm(const float* __restrict__ A,
                                              const float* __restrict__ W,
                                              const float* __restrict__ bias,
                                              float* __restrict__ C,
                                              float* __restrict__ gstats, int nRows) {
  __shared__ alignas(32) bf16_t Bs[4 * 8 * 32 * 16];   // 16384 bf16 = 32 KB
  __shared__ float sstats[2 * D];                      // col sums + sumsq

  int tid = threadIdx.x;
  if (tid < 2 * D) sstats[tid] = 0.0f;
  for (int f = tid; f < 16384; f += 256) {
    int e     =  f        & 15;
    int ln    = (f >> 4)  & 31;
    int ntile = (f >> 9)  & 7;
    int kstep = (f >> 12) & 3;
    int n = ntile * 16 + (ln & 15);
    int k = kstep * 32 + ((ln >> 4) * 16) + e;
    Bs[f] = f2bf(W[k * D + n]);
  }
  __syncthreads();

  int wave  = tid >> 5;
  int lane  = tid & 31;
  int row0  = blockIdx.x * 128 + wave * 16;
  bool active = (row0 < nRows);              // N is a multiple of 16

  if (active) {
    int m     = row0 + (lane & 15);
    int khalf = lane >> 4;

    v8f acc[8];
#pragma unroll
    for (int t = 0; t < 8; ++t) acc[t] = (v8f){0.f,0.f,0.f,0.f,0.f,0.f,0.f,0.f};

#pragma unroll
    for (int kstep = 0; kstep < 4; ++kstep) {
      // A fragment, contiguous order matching Bs: K = 16*khalf+e
      const float* rowp = A + (long)m * D + kstep * 32;
      float4 f0 = *(const float4*)(rowp + 16 * khalf);
      float4 f1 = *(const float4*)(rowp + 16 * khalf + 4);
      float4 f2 = *(const float4*)(rowp + 16 * khalf + 8);
      float4 f3 = *(const float4*)(rowp + 16 * khalf + 12);
      v16bf a;
      a[0]=f2bf(f0.x);  a[1]=f2bf(f0.y);  a[2]=f2bf(f0.z);  a[3]=f2bf(f0.w);
      a[4]=f2bf(f1.x);  a[5]=f2bf(f1.y);  a[6]=f2bf(f1.z);  a[7]=f2bf(f1.w);
      a[8]=f2bf(f2.x);  a[9]=f2bf(f2.y);  a[10]=f2bf(f2.z); a[11]=f2bf(f2.w);
      a[12]=f2bf(f3.x); a[13]=f2bf(f3.y); a[14]=f2bf(f3.z); a[15]=f2bf(f3.w);

#pragma unroll
      for (int nt = 0; nt < 8; ++nt) {
        v16bf b = *(const v16bf*)&Bs[((kstep * 8 + nt) * 32 + lane) * 16];
        acc[nt] = __builtin_amdgcn_wmma_f32_16x16x32_bf16(
            false, a, false, b, (short)0, acc[nt], false, false);
      }
    }

    // C/D layout: VGPR r -> row (lane>=16 ? 8 : 0)+r, col = lane&15 within tile
    int colBase = lane & 15;
    int rowOff  = (lane >> 4) * 8;
#pragma unroll
    for (int nt = 0; nt < 8; ++nt) {
      int col = nt * 16 + colBase;
      float bv = bias[col];
      float s = 0.f, s2 = 0.f;
#pragma unroll
      for (int r = 0; r < 8; ++r) {
        float v = acc[nt][r] + bv;
        C[(long)(row0 + rowOff + r) * D + col] = v;
        s += v; s2 += v * v;
      }
      atomicAdd(&sstats[col], s);
      atomicAdd(&sstats[D + col], s2);
    }
  }

  __syncthreads();
  atomicAdd(&gstats[tid], sstats[tid & (2 * D - 1)]);   // tid == index, 256 entries
}

// ------------------------------------------------- x = relu(bn(x)) in place,
// accumulating column stats of the OUTPUT for the next BatchNorm.
// 32 rows per block of 256 threads (2 thread-groups of 128 columns).
__global__ __launch_bounds__(256) void k_bnrelu_stats(float* __restrict__ x,
                                                      const float* __restrict__ stats_in,
                                                      const float* __restrict__ g,
                                                      const float* __restrict__ b,
                                                      float* __restrict__ stats_out,
                                                      int nRows) {
  __shared__ float ls[2 * D];
  int tid = threadIdx.x;
  if (tid < 2 * D) ls[tid] = 0.0f;
  __syncthreads();

  int c     = tid & (D - 1);
  int rbase = blockIdx.x * 32 + (tid >> 7) * 16;
  float invN = 1.0f / (float)nRows;
  float m   = stats_in[c] * invN;
  float var = stats_in[D + c] * invN - m * m;
  float inv = rsqrtf(var + BN_EPS);
  float gg = g[c], bb = b[c];

  float s = 0.f, s2 = 0.f;
#pragma unroll
  for (int i = 0; i < 16; ++i) {
    int r = rbase + i;
    if (r < nRows) {
      long idx = (long)r * D + c;
      float v = (x[idx] - m) * inv * gg + bb;
      v = v > 0.f ? v : 0.f;
      x[idx] = v;
      s += v; s2 += v * v;
    }
  }
  atomicAdd(&ls[c], s);
  atomicAdd(&ls[D + c], s2);

  __syncthreads();
  atomicAdd(&stats_out[tid], ls[tid & (2 * D - 1)]);
}

// ------------------------------------------------- x = relu(bn(x)) in place (no stats out)
__global__ void k_bnrelu(float* __restrict__ x, const float* __restrict__ stats,
                         const float* __restrict__ g, const float* __restrict__ b, int nRows) {
  long i = (long)blockIdx.x * blockDim.x + threadIdx.x;
  if (i >= (long)nRows * D) return;
  int c = (int)(i & (D - 1));
  float invN = 1.0f / (float)nRows;
  float m = stats[c] * invN;
  float var = stats[D + c] * invN - m * m;
  float inv = rsqrtf(var + BN_EPS);
  float v = (x[i] - m) * inv * g[c] + b[c];
  x[i] = v > 0.f ? v : 0.f;
}

// ------------------------------------------------- pooled[gid[n]] += h[n] ; wave per node
__global__ void k_pool(const float* __restrict__ h, const int* __restrict__ gid,
                       float* __restrict__ pooled, int nNodes) {
  int node = (int)(((long)blockIdx.x * blockDim.x + threadIdx.x) >> 5);
  int lane = threadIdx.x & 31;
  if (node >= nNodes) return;
  int g = gid[node];
  const float4* hp = (const float4*)(h + (long)node * D);
  float*        pp = pooled + (long)g * D;
  float4 v = hp[lane];
  atomicAdd(&pp[lane * 4 + 0], v.x);
  atomicAdd(&pp[lane * 4 + 1], v.y);
  atomicAdd(&pp[lane * 4 + 2], v.z);
  atomicAdd(&pp[lane * 4 + 3], v.w);
}

// ------------------------------------------------- score = sum_k pooled[k] @ W_k + b_k
__global__ void k_score(const float* __restrict__ pooled0, const float* __restrict__ pooledR,
                        const float* __restrict__ predW, const float* __restrict__ predb,
                        float* __restrict__ score) {
  int idx = blockIdx.x * blockDim.x + threadIdx.x;
  if (idx >= NGRAPH * OUTD) return;
  int g = idx >> 6, o = idx & 63;
  float acc = 0.f;
  for (int k = 0; k <= NLAYER; ++k) {
    const float* p = (k == 0) ? (pooled0 + g * D)
                              : (pooledR + (long)(k - 1) * NGRAPH * D + g * D);
    const float* w = predW + (long)k * D * OUTD + o;   // predW[k][d][o]
    float s = 0.f;
    for (int d = 0; d < D; ++d) s += p[d] * w[d * OUTD];
    acc += s + predb[k * OUTD + o];
  }
  score[idx] = acc;
}

// ================================================================ launch
extern "C" void kernel_launch(void* const* d_in, const int* in_sizes, int n_in,
                              void* d_out, int out_size, void* d_ws, size_t ws_size,
                              hipStream_t stream) {
  const float* h0    = (const float*)d_in[0];
  const int*   src   = (const int*)  d_in[2];
  const int*   dst   = (const int*)  d_in[3];
  const int*   gid   = (const int*)  d_in[4];
  const float* eps   = (const float*)d_in[5];
  const float* W1    = (const float*)d_in[6];
  const float* b1    = (const float*)d_in[7];
  const float* W2    = (const float*)d_in[8];
  const float* b2    = (const float*)d_in[9];
  const float* bn1g  = (const float*)d_in[10];
  const float* bn1b  = (const float*)d_in[11];
  const float* bn2g  = (const float*)d_in[12];
  const float* bn2b  = (const float*)d_in[13];
  const float* bn3g  = (const float*)d_in[14];
  const float* bn3b  = (const float*)d_in[15];
  const float* predW = (const float*)d_in[16];
  const float* predb = (const float*)d_in[17];

  const int  N  = in_sizes[0] / D;     // 100000
  const int  E  = in_sizes[2];         // 1600000
  const long ND = (long)N * D;

  float* out     = (float*)d_out;
  float* B1      = (float*)d_ws;       // agg / MLP input
  float* B2      = B1 + ND;            // ping
  float* B3      = B2 + ND;            // pong
  float* statsA  = B3 + ND;            // 2*D
  float* statsB  = statsA + 2 * D;     // 2*D
  float* pooled0 = statsB + 2 * D;     // G*D

  const int TB = 256;
  const long outN = (long)NGRAPH * OUTD + (long)NLAYER * NGRAPH * D;   // 36864

  // zero accumulation targets (d_out / pooled0 are poisoned by the harness)
  k_zero<<<(int)((outN + TB - 1) / TB), TB, 0, stream>>>(out, outN);
  k_zero<<<(NGRAPH * D + TB - 1) / TB, TB, 0, stream>>>(pooled0, NGRAPH * D);

  // pooled[0] = segment_sum(h0, graph_id)
  k_pool<<<(N + 7) / 8, TB, 0, stream>>>(h0, gid, pooled0, N);

  const float* hprev = h0;
  float* ping = B2;
  float* pong = B3;

  for (int i = 0; i < NLAYER; ++i) {
    // agg = segment_sum(h[src], dst)
    k_zero<<<(int)((ND + TB - 1) / TB), TB, 0, stream>>>(B1, ND);
    k_scatter<<<(E + 7) / 8, TB, 0, stream>>>(hprev, src, dst, B1, E);
    // x = (1+eps)*h + agg   (in place in B1)
    k_axpy<<<(int)((ND + TB - 1) / TB), TB, 0, stream>>>(B1, hprev, eps, i, ND);

    float* hout = ping;   // != hprev by construction

    // y = x @ W1 + b1  (epilogue accumulates column stats into statsA)
    k_zero<<<1, TB, 0, stream>>>(statsA, 2 * D);
    k_gemm<<<(N + 127) / 128, TB, 0, stream>>>(B1, W1 + (long)i * D * D, b1 + i * D,
                                               hout, statsA, N);
    // y = relu(bn(y))  (no stats out needed: next stats come from GEMM2 epilogue)
    k_bnrelu<<<(int)((ND + TB - 1) / TB), TB, 0, stream>>>(hout, statsA,
                                                           bn1g + i * D, bn1b + i * D, N);

    // z = y @ W2 + b2  (in-place: each wave reads only its own rows; stats -> statsA)
    k_zero<<<1, TB, 0, stream>>>(statsA, 2 * D);
    k_gemm<<<(N + 127) / 128, TB, 0, stream>>>(hout, W2 + (long)i * D * D, b2 + i * D,
                                               hout, statsA, N);

    // apply BN -> ReLU, fused stats of the output -> statsB
    k_zero<<<1, TB, 0, stream>>>(statsB, 2 * D);
    k_bnrelu_stats<<<(N + 31) / 32, TB, 0, stream>>>(hout, statsA,
                                                     bn2g + i * D, bn2b + i * D, statsB, N);
    // outer BN -> ReLU (no stats out)
    k_bnrelu<<<(int)((ND + TB - 1) / TB), TB, 0, stream>>>(hout, statsB,
                                                           bn3g + i * D, bn3b + i * D, N);

    // pooled[i+1] accumulates directly into d_out
    k_pool<<<(N + 7) / 8, TB, 0, stream>>>(hout, gid, out + NGRAPH * OUTD + (long)i * NGRAPH * D, N);

    hprev = hout;
    float* t = ping; ping = pong; pong = t;
  }

  // score = sum_k pooled[k] @ pred_W[k] + pred_b[k]
  k_score<<<(NGRAPH * OUTD + TB - 1) / TB, TB, 0, stream>>>(
      pooled0, out + NGRAPH * OUTD, predW, predb, out);
}
